// VectorQuantizer_81363860455950
// MI455X (gfx1250) — matrix operand order repaired and checked by reading, compiled, not run on hardware
//
#include <hip/hip_runtime.h>
#include <math.h>

typedef __attribute__((ext_vector_type(16))) __bf16 v16bf;
typedef __attribute__((ext_vector_type(8)))  __bf16 v8bf;
typedef __attribute__((ext_vector_type(8)))  float  v8f;

#define K_CODES 2048
#define DIM     256
#define NPIX    65536          // 64*32*32
#define QOUT_OFF 1             // d_out: [loss][quant 4194304][perp][idx 65536]
#define PERP_OFF 4194305
#define IDX_OFF  4194306

// ---------------- kernel 0: zero accumulators (must re-zero every call) ----
__global__ void vq_zero(float* __restrict__ counts, float* __restrict__ loss) {
    int i = blockIdx.x * 256 + threadIdx.x;
    if (i < K_CODES) counts[i] = 0.0f;
    if (i == 0) *loss = 0.0f;
}

// ---------------- kernel 1: split codebook -> bf16 hi/lo, code norms -------
__global__ __launch_bounds__(256)
void vq_prep(const float* __restrict__ cb, __bf16* __restrict__ cb_hi,
             __bf16* __restrict__ cb_lo, float* __restrict__ c_norm) {
    __shared__ float red[256];
    const int k = blockIdx.x, d = threadIdx.x;
    float c = cb[(size_t)k * DIM + d];
    __bf16 h = (__bf16)c;
    cb_hi[(size_t)k * DIM + d] = h;
    cb_lo[(size_t)k * DIM + d] = (__bf16)(c - (float)h);
    red[d] = c * c;
    __syncthreads();
    for (int off = 128; off; off >>= 1) {
        if (d < off) red[d] += red[d + off];
        __syncthreads();
    }
    if (d == 0) c_norm[k] = red[0];
}

// ---------------- kernel 2: WMMA distance GEMM + argmin --------------------
// 512 blocks x 256 threads (8 waves). Wave w owns pixels [blk*128+16w, +16).
// score[n,k] = ||c_k||^2 - 2 * x_n . c_k   (||x||^2 constant per row).
__global__ __launch_bounds__(256)
void vq_main(const float* __restrict__ in, const __bf16* __restrict__ cb_hi,
             const __bf16* __restrict__ cb_lo, const float* __restrict__ c_norm,
             int* __restrict__ idx_ws) {
    __shared__ float s_norm[K_CODES];
    for (int i = threadIdx.x; i < K_CODES; i += 256) s_norm[i] = c_norm[i];
    __syncthreads();

    const int lane = threadIdx.x & 31;
    const int wave = threadIdx.x >> 5;
    const int lh   = lane >> 4;      // which 16-lane half
    const int ln   = lane & 15;
    const int pix  = blockIdx.x * 128 + wave * 16 + ln;
    const int bB   = pix >> 10;      // batch
    const int hw   = pix & 1023;     // h*32+w
    const float* __restrict__ px = in + (size_t)bB * 262144 + hw;  // px[d*1024]

    // --- Build resident B fragments (hi+lo) for this wave's 16 pixels.
    // B layout (32x16, 16-bit): lane = column(pixel), element e -> K = e + 16*lh.
    v16bf bh[8], bl[8];
#pragma unroll
    for (int c = 0; c < 8; ++c) {
        const int dbase = c * 32 + lh * 16;
        v16bf h, l;
#pragma unroll
        for (int e = 0; e < 16; ++e) {
            float x  = px[(size_t)(dbase + e) * 1024];
            __bf16 xh = (__bf16)x;
            h[e] = xh;
            l[e] = (__bf16)(x - (float)xh);
        }
        bh[c] = h; bl[c] = l;
    }

    float minv = 3.4e38f;
    int   mini = 0;

    for (int t = 0; t < 128; ++t) {
        const int k0 = t * 16 + ln;  // A row (code) for this lane
        const __bf16* __restrict__ rh = cb_hi + (size_t)k0 * DIM;
        const __bf16* __restrict__ rl = cb_lo + (size_t)k0 * DIM;
        if (t + 1 < 128) {
            __builtin_prefetch(cb_hi + (size_t)(k0 + 16) * DIM, 0, 1);
            __builtin_prefetch(cb_lo + (size_t)(k0 + 16) * DIM, 0, 1);
        }
        v8f acc = {};
#pragma unroll
        for (int c = 0; c < 8; ++c) {
            // A layout (16x32, 16-bit): element e -> K = base+lh*16+e.
            const int off = c * 32 + lh * 16;
            v8bf a0 = *(const v8bf*)(rh + off);
            v8bf a1 = *(const v8bf*)(rh + off + 8);
            v8bf c0 = *(const v8bf*)(rl + off);
            v8bf c1 = *(const v8bf*)(rl + off + 8);
            v16bf ah = __builtin_shufflevector(a0, a1, 0,1,2,3,4,5,6,7,8,9,10,11,12,13,14,15);
            v16bf al = __builtin_shufflevector(c0, c1, 0,1,2,3,4,5,6,7,8,9,10,11,12,13,14,15);
            acc = __builtin_amdgcn_wmma_f32_16x16x32_bf16(false, ah, false, bh[c], (short)0, acc, false, false);
            acc = __builtin_amdgcn_wmma_f32_16x16x32_bf16(false, ah, false, bl[c], (short)0, acc, false, false);
            acc = __builtin_amdgcn_wmma_f32_16x16x32_bf16(false, al, false, bh[c], (short)0, acc, false, false);
        }
        // D layout: lane l, vgpr r -> code M = t*16 + r + 8*lh, pixel N = l&15.
#pragma unroll
        for (int r = 0; r < 8; ++r) {
            const int code = t * 16 + lh * 8 + r;
            float s = s_norm[code] - 2.0f * acc[r];
            if (s < minv) { minv = s; mini = code; }  // ascending scan: first-min kept
        }
    }

    // Combine the two 16-lane halves (each saw a disjoint half of the codes).
    float ov = __shfl_xor(minv, 16, 32);
    int   oi = __shfl_xor(mini, 16, 32);
    if (ov < minv || (ov == minv && oi < mini)) { minv = ov; mini = oi; }
    if (lh == 0) idx_ws[pix] = mini;
}

// ---------------- kernel 3: gather codes, write output, loss, counts, idx --
__global__ __launch_bounds__(256)
void vq_gather(const float* __restrict__ in, const float* __restrict__ cb,
               const int* __restrict__ idx_ws, float* __restrict__ counts,
               float* __restrict__ loss_acc, float* __restrict__ out) {
    __shared__ float red[256];
    const int t   = threadIdx.x;
    const int pl  = t & 127;
    const int sub = t >> 7;
    const int pix = blockIdx.x * 128 + pl;
    const int bB  = pix >> 10;
    const int hw  = pix & 1023;
    const int idx = idx_ws[pix];
    const float* __restrict__ crow = cb + (size_t)idx * DIM;
    const float* __restrict__ px   = in + (size_t)bB * 262144 + hw;
    float* __restrict__ qo = out + QOUT_OFF + (size_t)bB * 262144 + hw;

    float s = 0.0f;
    for (int d = sub; d < DIM; d += 2) {
        float q = crow[d];
        float x = px[(size_t)d * 1024];
        qo[(size_t)d * 1024] = q;   // straight-through output == quant numerically
        float df = q - x;
        s += df * df;
    }
    red[t] = s;
    __syncthreads();
    for (int off = 128; off; off >>= 1) {
        if (t < off) red[t] += red[t + off];
        __syncthreads();
    }
    if (t == 0) atomicAdd(loss_acc, red[0]);
    if (sub == 0) {
        atomicAdd(&counts[idx], 1.0f);
        out[IDX_OFF + pix] = (float)idx;
    }
}

// ---------------- kernel 4: loss scalar + perplexity -----------------------
__global__ __launch_bounds__(256)
void vq_final(const float* __restrict__ counts, const float* __restrict__ loss_acc,
              float* __restrict__ out) {
    __shared__ float red[256];
    const int t = threadIdx.x;
    float h = 0.0f;
    for (int k = t; k < K_CODES; k += 256) {
        float p = counts[k] * (1.0f / (float)NPIX);
        h += p * logf(p + 1e-10f);
    }
    red[t] = h;
    __syncthreads();
    for (int off = 128; off; off >>= 1) {
        if (t < off) red[t] += red[t + off];
        __syncthreads();
    }
    if (t == 0) {
        out[PERP_OFF] = expf(-red[0]);
        // loss = q_latent + cc*e_latent = (1+0.25)*mean((q-x)^2)
        out[0] = (*loss_acc) * (1.25f / ((float)NPIX * (float)DIM));
    }
}

// ---------------- host launch ----------------------------------------------
extern "C" void kernel_launch(void* const* d_in, const int* in_sizes, int n_in,
                              void* d_out, int out_size, void* d_ws, size_t ws_size,
                              hipStream_t stream) {
    const float* in = (const float*)d_in[0];   // [64,256,32,32]
    const float* cb = (const float*)d_in[1];   // [2048,256]
    float* out = (float*)d_out;
    char*  ws  = (char*)d_ws;

    __bf16* cb_hi  = (__bf16*)(ws);                       // 1 MB
    __bf16* cb_lo  = (__bf16*)(ws + (1u << 20));          // 1 MB
    float*  c_norm = (float*)(ws + (2u << 20));           // 8 KB
    float*  counts = (float*)(ws + (2u << 20) + 8192);    // 8 KB
    float*  loss   = (float*)(ws + (2u << 20) + 16384);   // 4 B
    int*    idx_ws = (int*)(ws + (2u << 20) + 16384 + 256); // 256 KB

    vq_zero  <<<8,    256, 0, stream>>>(counts, loss);
    vq_prep  <<<2048, 256, 0, stream>>>(cb, cb_hi, cb_lo, c_norm);
    vq_main  <<<512,  256, 0, stream>>>(in, cb_hi, cb_lo, c_norm, idx_ws);
    vq_gather<<<512,  256, 0, stream>>>(in, cb, idx_ws, counts, loss, out);
    vq_final <<<1,    256, 0, stream>>>(counts, loss, out);
}
